// SparseAutoencoderBase_72911364817359
// MI455X (gfx1250) — compile-verified, matrix operand order checked by reading, not run on hardware
//
#include <hip/hip_runtime.h>
#include <hip/hip_bf16.h>

// Problem constants (fixed by the reference):
//   x: [B=8192, D=768] f32, W: [D=768, L=16384] f32, b_enc: [L], b_dec: [D]
//   z = x@W + b_enc; top-32 by |z| per row -> z_sparse; recon = z_sparse@W.T + b_dec
//   d_out = [recon (B*D) | z_sparse (B*L)] f32
static constexpr int BDIM = 8192;
static constexpr int DDIM = 768;
static constexpr int LDIM = 16384;
static constexpr int KSEL = 32;

typedef __attribute__((ext_vector_type(16))) __bf16 v16bf;
typedef __attribute__((ext_vector_type(8)))  float  v8f;

union FragBF  { uint4 u[2]; v16bf v; };
union PackBF4 { __bf16 h[4]; uint2 u2; };
union PackBF2 { __bf16 h[2]; unsigned int u; };

// ---------------------------------------------------------------------------
// Kernel 1: z = x @ W + b_enc  via V_WMMA_F32_16X16X32_BF16
//
// Workgroup: 256 threads (8 wave32), output tile 128(M) x 256(N), K-step 32.
// Wave grid 2x4: wave (wr,wc) owns rows [wr*64,+64) x cols [wc*64,+64)
//   -> 4x4 tiles of 16x16x32, accumulators v8f acc[4][4] (128 VGPRs).
// Double-buffered LDS + register-staged global prefetch: global loads for
// K-step k+1 are issued right after the (single) barrier of step k and
// overlap the 128 WMMAs of step k.
// LDS layout [row|col][K] with padded stride 40 bf16 (80 B): per-lane b128
// fragment reads hit distinct bank groups (stride 20 banks, period 16).
// B is packed K-pairwise ((bf16 k, bf16 k+1) per dword) so the transposed
// staging uses b32 stores instead of b16 scatter.
// ---------------------------------------------------------------------------
__global__ __launch_bounds__(256)
void sae_encode_wmma(const float* __restrict__ x,
                     const float* __restrict__ W,
                     const float* __restrict__ b_enc,
                     float* __restrict__ z)
{
    __shared__ __bf16 lA[2][128][40];   // 20,480 B
    __shared__ __bf16 lB[2][256][40];   // 40,960 B

    const int tid  = threadIdx.x;
    const int lane = tid & 31;
    const int wave = tid >> 5;
    const int wr   = wave >> 2;          // 0..1  (64 rows each)
    const int wc   = wave & 3;           // 0..3  (64 cols each)
    const int l16  = lane & 15;
    const int hi   = lane >> 4;          // selects K-half per WMMA layout
    const int m0   = blockIdx.y * 128;
    const int n0   = blockIdx.x * 256;

    // Per-thread staging coordinates (constant across K-steps).
    const int ar = (tid * 4 + 0) >> 5;          // unused placeholder pattern
    (void)ar;

    v8f acc[4][4] = {};

    float4 ra[4];            // A stage: 4096 f32 / 256 thr = 4 float4
    float4 rb0[4], rb1[4];   // B stage: 8192 f32 -> 4 K-pair float4 pairs

    // --- global tile prefetch into registers ---
    auto load_tiles = [&](int kk) {
#pragma unroll
        for (int i = 0; i < 4; ++i) {
            const int j = tid + i * 256;        // float4 idx 0..1023
            const int r = j >> 3;               // A row 0..127
            const int c = (j & 7) * 4;          // A col 0..28
            ra[i] = *reinterpret_cast<const float4*>(
                &x[(size_t)(m0 + r) * DDIM + kk + c]);
        }
#pragma unroll
        for (int i = 0; i < 4; ++i) {
            const int p  = tid + i * 256;       // pair-group 0..1023
            const int k  = (p >> 6) * 2;        // even k 0..30
            const int nq = (p & 63) * 4;        // n 0..252
            rb0[i] = *reinterpret_cast<const float4*>(
                &W[(size_t)(kk + k) * LDIM + n0 + nq]);
            rb1[i] = *reinterpret_cast<const float4*>(
                &W[(size_t)(kk + k + 1) * LDIM + n0 + nq]);
        }
    };

    load_tiles(0);

    for (int kk = 0; kk < DDIM; kk += 32) {
        const int buf = (kk >> 5) & 1;

        // --- commit staged registers to LDS (bf16, packed stores) ---
#pragma unroll
        for (int i = 0; i < 4; ++i) {
            const int j = tid + i * 256;
            const int r = j >> 3;
            const int c = (j & 7) * 4;
            PackBF4 pk;
            pk.h[0] = (__bf16)ra[i].x;  pk.h[1] = (__bf16)ra[i].y;
            pk.h[2] = (__bf16)ra[i].z;  pk.h[3] = (__bf16)ra[i].w;
            *reinterpret_cast<uint2*>(&lA[buf][r][c]) = pk.u2;
        }
#pragma unroll
        for (int i = 0; i < 4; ++i) {
            const int p  = tid + i * 256;
            const int k  = (p >> 6) * 2;
            const int nq = (p & 63) * 4;
            const float f0[4] = { rb0[i].x, rb0[i].y, rb0[i].z, rb0[i].w };
            const float f1[4] = { rb1[i].x, rb1[i].y, rb1[i].z, rb1[i].w };
#pragma unroll
            for (int q = 0; q < 4; ++q) {
                PackBF2 pk;
                pk.h[0] = (__bf16)f0[q];        // K = k   -> bits [15:0]
                pk.h[1] = (__bf16)f1[q];        // K = k+1 -> bits [31:16]
                *reinterpret_cast<unsigned int*>(&lB[buf][nq + q][k]) = pk.u;
            }
        }
        __syncthreads();

        // --- prefetch next K-step while this one computes ---
        if (kk + 32 < DDIM) load_tiles(kk + 32);

        // --- fragment loads (documented 16-bit A layout):
        // lanes 0-15: K0-7 then K16-23; lanes 16-31: K8-15 then K24-31 ---
        FragBF a[4], b[4];
#pragma unroll
        for (int i = 0; i < 4; ++i) {
            const __bf16* p = &lA[buf][wr * 64 + i * 16 + l16][hi * 8];
            a[i].u[0] = *reinterpret_cast<const uint4*>(p);
            a[i].u[1] = *reinterpret_cast<const uint4*>(p + 16);
        }
#pragma unroll
        for (int jn = 0; jn < 4; ++jn) {
            const __bf16* p = &lB[buf][wc * 64 + jn * 16 + l16][hi * 8];
            b[jn].u[0] = *reinterpret_cast<const uint4*>(p);
            b[jn].u[1] = *reinterpret_cast<const uint4*>(p + 16);
        }

#pragma unroll
        for (int i = 0; i < 4; ++i)
#pragma unroll
            for (int jn = 0; jn < 4; ++jn)
                acc[i][jn] = __builtin_amdgcn_wmma_f32_16x16x32_bf16(
                    false, a[i].v, false, b[jn].v,
                    (short)0, acc[i][jn], false, false);
        // single barrier per K-step: the next iteration's barrier separates
        // this step's reads from the write that reuses this buffer (i+2).
    }

    // --- epilogue: add b_enc, store dense z (scratch for top-k pass) ---
    // C layout: VGPR r -> M = r (lanes 0-15) / 8+r (lanes 16-31), N = lane%16.
#pragma unroll
    for (int jn = 0; jn < 4; ++jn) {
        const int   n  = n0 + wc * 64 + jn * 16 + l16;
        const float be = b_enc[n];
#pragma unroll
        for (int i = 0; i < 4; ++i) {
            const int mb = m0 + wr * 64 + i * 16 + hi * 8;
#pragma unroll
            for (int r = 0; r < 8; ++r)
                z[(size_t)(mb + r) * LDIM + n] = acc[i][jn][r] + be;
        }
    }
}

// ---------------------------------------------------------------------------
// Kernel 2: per-row exact top-32 radix select on |z| bits, sparsify row in
// place, then sparse decode via gathers from L2-resident W.
// One workgroup (256 threads, 8 waves) per row; all row sweeps use float4.
// ---------------------------------------------------------------------------
__global__ __launch_bounds__(256)
void sae_topk_decode(const float* __restrict__ W,
                     const float* __restrict__ b_dec,
                     float* __restrict__ z,
                     float* __restrict__ recon)
{
    __shared__ unsigned int hist[8][256];     // per-wave sub-histograms
    __shared__ unsigned int s_selbin, s_remaining, s_eq, s_cnt;
    __shared__ int   sidx[64];
    __shared__ float sval[64];

    const int tid  = threadIdx.x;
    const int wave = tid >> 5;
    const int row  = blockIdx.x;
    float* zrow = z + (size_t)row * LDIM;
    const float4* zrow4 = reinterpret_cast<const float4*>(zrow);

    unsigned int prefix    = 0;
    unsigned int remaining = KSEL;

    // 4-pass MSB-first radix select: |f32| bit pattern order == uint order.
    for (int shift = 24; shift >= 0; shift -= 8) {
        for (int i = tid; i < 8 * 256; i += 256)
            (&hist[0][0])[i] = 0;
        __syncthreads();

        unsigned int* myhist = hist[wave];
        for (int i = tid; i < LDIM / 4; i += 256) {
            const float4 f = zrow4[i];
            const float e[4] = { f.x, f.y, f.z, f.w };
#pragma unroll
            for (int q = 0; q < 4; ++q) {
                const unsigned int v = __float_as_uint(e[q]) & 0x7FFFFFFFu;
                const bool match =
                    (shift == 24) || ((v >> (shift + 8)) == prefix);
                if (match) atomicAdd(&myhist[(v >> shift) & 0xFFu], 1u);
            }
        }
        __syncthreads();

        if (tid == 0) {
            unsigned int running = 0;
            int sel = 0;
            unsigned int rem = remaining;
            for (int b = 255; b >= 0; --b) {
                unsigned int c = 0;
#pragma unroll
                for (int w = 0; w < 8; ++w) c += hist[w][b];
                if (running + c >= remaining) { sel = b; rem = remaining - running; break; }
                running += c;
            }
            s_selbin = (unsigned int)sel;
            s_remaining = rem;
        }
        __syncthreads();
        prefix    = (prefix << 8) | s_selbin;
        remaining = s_remaining;
        __syncthreads();
    }
    const unsigned int thresh     = prefix;      // K-th largest |z| bits
    const unsigned int tiesNeeded = remaining;   // # of == thresh to keep

    if (tid == 0) { s_eq = 0; s_cnt = 0; }
    __syncthreads();

    // Sparsify row in place (float4 traffic) + compact selected pairs.
    for (int i = tid; i < LDIM / 4; i += 256) {
        float4 f = zrow4[i];
        float e[4] = { f.x, f.y, f.z, f.w };
#pragma unroll
        for (int q = 0; q < 4; ++q) {
            const unsigned int v = __float_as_uint(e[q]) & 0x7FFFFFFFu;
            bool keep = (v > thresh);
            if (!keep && v == thresh)
                keep = (atomicAdd(&s_eq, 1u) < tiesNeeded);
            if (keep) {
                const unsigned int s = atomicAdd(&s_cnt, 1u);
                if (s < 64u) { sidx[s] = i * 4 + q; sval[s] = e[q]; }
            } else {
                e[q] = 0.0f;
            }
        }
        f.x = e[0]; f.y = e[1]; f.z = e[2]; f.w = e[3];
        reinterpret_cast<float4*>(zrow)[i] = f;   // final z_sparse output
    }
    __syncthreads();

    const unsigned int nsel = (s_cnt < 64u) ? s_cnt : 64u;

    // Sparse decode: recon[row, d] = b_dec[d] + sum_k sval[k] * W[d, sidx[k]]
    for (int d = tid; d < DDIM; d += 256) {
        float acc = b_dec[d];
        const float* wrow = W + (size_t)d * LDIM;
        for (unsigned int k = 0; k < nsel; ++k)
            acc += sval[k] * wrow[sidx[k]];
        recon[(size_t)row * DDIM + d] = acc;
    }
}

// ---------------------------------------------------------------------------
extern "C" void kernel_launch(void* const* d_in, const int* in_sizes, int n_in,
                              void* d_out, int out_size, void* d_ws, size_t ws_size,
                              hipStream_t stream)
{
    (void)in_sizes; (void)n_in; (void)out_size; (void)d_ws; (void)ws_size;

    const float* x     = (const float*)d_in[0];
    const float* W     = (const float*)d_in[1];
    const float* b_enc = (const float*)d_in[2];
    const float* b_dec = (const float*)d_in[3];

    float* recon = (float*)d_out;                       // [B, D]
    float* z     = recon + (size_t)BDIM * DDIM;         // [B, L] (dense then sparse)

    dim3 grid1(LDIM / 256, BDIM / 128);                 // 64 x 64 workgroups
    sae_encode_wmma<<<grid1, 256, 0, stream>>>(x, W, b_enc, z);

    sae_topk_decode<<<BDIM, 256, 0, stream>>>(W, b_dec, z, recon);
}
